// RA_MLA_Attention_73478300500273
// MI455X (gfx1250) — matrix operand order repaired and checked by reading, not compiled, not run on hardware
//
#include <hip/hip_runtime.h>

#define TT 2048
#define EE 2048
#define HH 16
#define DD 128
#define LL 64

typedef __bf16 v16bf __attribute__((ext_vector_type(16)));
typedef float  v8f   __attribute__((ext_vector_type(8)));

union Frag { v16bf v; uint4 q[2]; };

// ---- WMMA fragment loaders (NT GEMM: A[M,K] row-major, B[N,K] row-major) ----
// A 16x32 bf16 layout: lane m=lane&15, hi=lane>>4; elems 0..7 = K hi*8+0..7,
// elems 8..15 = K 16+hi*8+0..7  -> two contiguous 16B loads.
__device__ __forceinline__ v16bf frag_a(const __bf16* A, int lda, int row0, int k0, int lane) {
    int m = lane & 15, hi = lane >> 4;
    const __bf16* p = A + (size_t)(row0 + m) * lda + k0 + hi * 8;
    Frag f;
    f.q[0] = *(const uint4*)p;
    f.q[1] = *(const uint4*)(p + 16);
    return f.v;
}
// B 32x16 bf16 layout: lane n=lane&15, hi=lane>>4; elems 0..7 = K hi*8+0..7,
// elems 8..15 = K 16+hi*8+0..7  -> two contiguous 16B loads.
__device__ __forceinline__ v16bf frag_b(const __bf16* B, int ldb, int col0, int k0, int lane) {
    int n = lane & 15, hi = lane >> 4;
    const __bf16* p = B + (size_t)(col0 + n) * ldb + k0 + hi * 8;
    Frag f;
    f.q[0] = *(const uint4*)p;
    f.q[1] = *(const uint4*)(p + 16);
    return f.v;
}

__device__ __forceinline__ v8f wmma_bf16(v16bf a, v16bf b, v8f c) {
    return __builtin_amdgcn_wmma_f32_16x16x32_bf16(false, a, false, b, (short)0, c, false, false);
}

// ---------------- conversion kernels ----------------
__global__ __launch_bounds__(256) void cvt_f32_bf16(const float* __restrict__ s,
                                                    __bf16* __restrict__ d, int n) {
    int i = blockIdx.x * blockDim.x + threadIdx.x;
    if (i < n) d[i] = (__bf16)s[i];
}

// q_to_latent [H,D,L] -> [H,L,D] bf16
__global__ __launch_bounds__(256) void cvt_qtl(const float* __restrict__ s, __bf16* __restrict__ d) {
    int i = blockIdx.x * blockDim.x + threadIdx.x;
    if (i >= HH * DD * LL) return;
    int h = i / (DD * LL), rem = i % (DD * LL);
    int dd = rem / LL, l = rem % LL;
    d[(size_t)h * LL * DD + (size_t)l * DD + dd] = (__bf16)s[i];
}

// v_up [H,L,D] -> [H,D,L] bf16
__global__ __launch_bounds__(256) void cvt_vup(const float* __restrict__ s, __bf16* __restrict__ d) {
    int i = blockIdx.x * blockDim.x + threadIdx.x;
    if (i >= HH * LL * DD) return;
    int h = i / (LL * DD), rem = i % (LL * DD);
    int l = rem / DD, dd = rem % DD;
    d[(size_t)h * DD * LL + (size_t)dd * LL + l] = (__bf16)s[i];
}

// ---------------- generic batched NT GEMM (bf16 in, f32 acc) ----------------
// C[m,n] = sum_k A[m,k]*B[n,k]; per-wave 64x32 tile (8 WMMA / 12 b128-loads
// per 32-K step); store bf16 or f32, optionally transposed (Ct[n*ldc+m]).
__global__ __launch_bounds__(256) void gemm_nt_bf16(
    int M, int N, int K,
    const __bf16* __restrict__ A, int lda, long long sA,
    const __bf16* __restrict__ B, int ldb, long long sB,
    float* __restrict__ Cf, __bf16* __restrict__ Ch, int ldc, long long sC, int transC)
{
    int lane = threadIdx.x & 31;
    int w    = threadIdx.x >> 5;
    int bat  = blockIdx.y;
    A += (size_t)sA * bat;
    B += (size_t)sB * bat;
    if (Cf) Cf += (size_t)sC * bat;
    if (Ch) Ch += (size_t)sC * bat;

    int tilesN = (N + 31) >> 5;
    int tilesM = (M + 63) >> 6;
    long long gw = (long long)blockIdx.x * (blockDim.x >> 5) + w;
    int tn = (int)(gw % tilesN);
    int tm = (int)(gw / tilesN);
    if (tm >= tilesM) return;
    int row0 = tm * 64, col0 = tn * 32;

    v8f acc[4][2];
#pragma unroll
    for (int i = 0; i < 4; ++i)
#pragma unroll
        for (int j = 0; j < 2; ++j) acc[i][j] = v8f{};

    for (int k0 = 0; k0 < K; k0 += 32) {
        v16bf b0 = frag_b(B, ldb, col0,      k0, lane);
        v16bf b1 = frag_b(B, ldb, col0 + 16, k0, lane);
#pragma unroll
        for (int i = 0; i < 4; ++i) {
            v16bf a = frag_a(A, lda, row0 + i * 16, k0, lane);
            acc[i][0] = wmma_bf16(a, b0, acc[i][0]);
            acc[i][1] = wmma_bf16(a, b1, acc[i][1]);
        }
    }

    int n = lane & 15, hi = lane >> 4;
#pragma unroll
    for (int i = 0; i < 4; ++i) {
#pragma unroll
        for (int r = 0; r < 8; ++r) {
            int m  = row0 + i * 16 + r + hi * 8;
            int n0 = col0 + n;
            int n1 = n0 + 16;
            size_t i0 = transC ? (size_t)n0 * ldc + m : (size_t)m * ldc + n0;
            size_t i1 = transC ? (size_t)n1 * ldc + m : (size_t)m * ldc + n1;
            if (Ch) {
                Ch[i0] = (__bf16)acc[i][0][r];
                Ch[i1] = (__bf16)acc[i][1][r];
            } else {
                Cf[i0] = acc[i][0][r];
                Cf[i1] = acc[i][1][r];
            }
        }
    }
}

// ---------------- row sums for the reciprocal band term ----------------
__global__ __launch_bounds__(256) void rowsums(const __bf16* __restrict__ qlat,
                                               const __bf16* __restrict__ lk,
                                               float* __restrict__ qsum,
                                               float* __restrict__ ksum)
{
    int i = blockIdx.x * blockDim.x + threadIdx.x;
    if (i < TT * HH) {
        int t = i >> 4, h = i & 15;
        float s = 0.f;
#pragma unroll
        for (int l = 0; l < LL; ++l) s += (float)qlat[(size_t)t * (HH * LL) + h * LL + l];
        qsum[i] = s;
    }
    if (i < TT) {
        float s = 0.f;
#pragma unroll
        for (int l = 0; l < LL; ++l) s += (float)lk[(size_t)i * LL + l];
        ksum[i] = s;
    }
}

// ---------------- flash attention over latent space ----------------
// One wave per (head, 16-row query block). S = q_latent x latent_k^T (K=64),
// causal mask + band(|t-s|<=64): += 0.5*qsum[t]*ksum[s]/sqrt(L),
// online softmax, O += P x V_exp^T (Vt is [H,D,T] so K=s is contiguous).
__global__ __launch_bounds__(128) void attn_kernel(
    const __bf16* __restrict__ qlat,  // [T, H*L]
    const __bf16* __restrict__ lk,    // [T, L]
    const __bf16* __restrict__ vt,    // [H, D, T]
    const float*  __restrict__ qsum,  // [T, H]
    const float*  __restrict__ ksum,  // [T]
    __bf16* __restrict__ ctx)         // [T, H*D]
{
    __shared__ alignas(16) __bf16 plds[4][16 * 32];
    int lane = threadIdx.x & 31;
    int w    = threadIdx.x >> 5;
    int h    = blockIdx.y;
    int qb   = blockIdx.x * 4 + w;
    int t0   = qb * 16;
    int n = lane & 15, hi = lane >> 4;
    const float inv_sqrt_L = 0.125f;           // 1/sqrt(64)
    const float alpha_s    = 0.5f * 0.125f;    // RA_ALPHA / sqrt(L)

    // q_latent A-frags for this 16-row block (K = L = 64 -> 2 frags), loaded once
    v16bf aq0 = frag_a(qlat + h * LL, HH * LL, t0, 0,  lane);
    v16bf aq1 = frag_a(qlat + h * LL, HH * LL, t0, 32, lane);

    float mrow[8], lrow[8], qs[8];
    v8f o[8];
#pragma unroll
    for (int r = 0; r < 8; ++r) {
        mrow[r] = -1e30f;
        lrow[r] = 0.f;
        o[r] = v8f{};
        qs[r] = qsum[(size_t)(t0 + r + hi * 8) * HH + h];
    }

    const __bf16* vhead = vt + (size_t)h * DD * TT;
    int schunks = (t0 + 16 + 31) >> 5;
    for (int c = 0; c < schunks; ++c) {
        int s0 = c * 32;
        // S tiles: columns [s0, s0+16) and [s0+16, s0+32)
        v16bf bk00 = frag_b(lk, LL, s0,      0,  lane);
        v16bf bk01 = frag_b(lk, LL, s0,      32, lane);
        v16bf bk10 = frag_b(lk, LL, s0 + 16, 0,  lane);
        v16bf bk11 = frag_b(lk, LL, s0 + 16, 32, lane);
        v8f sA{}, sB{};
        sA = wmma_bf16(aq0, bk00, sA);
        sA = wmma_bf16(aq1, bk01, sA);
        sB = wmma_bf16(aq0, bk10, sB);
        sB = wmma_bf16(aq1, bk11, sB);

        int   sa  = s0 + n;
        int   sb  = sa + 16;
        float ksa = ksum[sa];
        float ksb = ksum[sb];

#pragma unroll
        for (int r = 0; r < 8; ++r) {
            int t = t0 + r + hi * 8;
            float va = sA[r] * inv_sqrt_L;
            float vb = sB[r] * inv_sqrt_L;
            bool ca = (sa <= t), cb = (sb <= t);
            if (ca && (t - sa <= 64)) va += alpha_s * qs[r] * ksa;
            if (cb && (t - sb <= 64)) vb += alpha_s * qs[r] * ksb;
            va = ca ? va : -1e30f;
            vb = cb ? vb : -1e30f;

            // row reduce over the 16 lanes holding this row (masks<16 stay in half)
            float mloc = fmaxf(va, vb);
#pragma unroll
            for (int off = 1; off < 16; off <<= 1)
                mloc = fmaxf(mloc, __shfl_xor(mloc, off, 32));
            float mnew  = fmaxf(mrow[r], mloc);
            float scale = __expf(mrow[r] - mnew);
            float pa = __expf(va - mnew);
            float pb = __expf(vb - mnew);
            float rs = pa + pb;
#pragma unroll
            for (int off = 1; off < 16; off <<= 1)
                rs += __shfl_xor(rs, off, 32);
            lrow[r] = lrow[r] * scale + rs;
            mrow[r] = mnew;
#pragma unroll
            for (int dti = 0; dti < 8; ++dti)
                o[dti][r] = o[dti][r] * scale;
            // stage P (C-layout) into LDS for A-layout reload
            plds[w][(r + hi * 8) * 32 + n]      = (__bf16)pa;
            plds[w][(r + hi * 8) * 32 + 16 + n] = (__bf16)pb;
        }
        asm volatile("s_wait_dscnt 0" ::: "memory");

        // reload P as a 16x32 A-fragment
        Frag fp;
        const __bf16* pp = &plds[w][(lane & 15) * 32 + hi * 8];
        fp.q[0] = *(const uint4*)pp;
        fp.q[1] = *(const uint4*)(pp + 16);

        // O[16x128] += P[16x32] x V[32x128]
#pragma unroll
        for (int dti = 0; dti < 8; ++dti) {
            v16bf bv = frag_b(vhead, TT, dti * 16, s0, lane);
            o[dti] = wmma_bf16(fp.v, bv, o[dti]);
        }
    }

    // normalize and write ctx [T, H*D] bf16
#pragma unroll
    for (int r = 0; r < 8; ++r) {
        float inv = 1.0f / lrow[r];
        size_t row = (size_t)(t0 + r + hi * 8) * (HH * DD) + h * DD;
#pragma unroll
        for (int dti = 0; dti < 8; ++dti)
            ctx[row + dti * 16 + n] = (__bf16)(o[dti][r] * inv);
    }
}

// ---------------- host launcher ----------------
extern "C" void kernel_launch(void* const* d_in, const int* in_sizes, int n_in,
                              void* d_out, int out_size, void* d_ws, size_t ws_size,
                              hipStream_t stream) {
    const float* hid = (const float*)d_in[0];
    const float* Wq  = (const float*)d_in[1];
    const float* Wk  = (const float*)d_in[2];
    const float* Wv  = (const float*)d_in[3];
    const float* q2l = (const float*)d_in[4];
    const float* vup = (const float*)d_in[5];
    const float* Wo  = (const float*)d_in[6];
    float* out = (float*)d_out;

    char* base = (char*)d_ws;
    size_t off = 0;
    auto alloc = [&](size_t elems, size_t esz) -> void* {
        void* p = base + off;
        off = (off + elems * esz + 255) & ~(size_t)255;
        return p;
    };
    __bf16* hid_h  = (__bf16*)alloc((size_t)TT * EE, 2);
    __bf16* Wq_h   = (__bf16*)alloc((size_t)EE * EE, 2);
    __bf16* Wk_h   = (__bf16*)alloc((size_t)LL * EE, 2);
    __bf16* Wv_h   = (__bf16*)alloc((size_t)LL * EE, 2);
    __bf16* Wo_h   = (__bf16*)alloc((size_t)EE * EE, 2);
    __bf16* qtl_t  = (__bf16*)alloc((size_t)HH * LL * DD, 2);
    __bf16* vup_t  = (__bf16*)alloc((size_t)HH * DD * LL, 2);
    __bf16* Q_h    = (__bf16*)alloc((size_t)TT * EE, 2);
    __bf16* lk_h   = (__bf16*)alloc((size_t)TT * LL, 2);
    __bf16* lv_h   = (__bf16*)alloc((size_t)TT * LL, 2);
    __bf16* qlat_h = (__bf16*)alloc((size_t)TT * HH * LL, 2);
    __bf16* Vt_h   = (__bf16*)alloc((size_t)HH * DD * TT, 2);
    __bf16* ctx_h  = (__bf16*)alloc((size_t)TT * EE, 2);
    float*  qsum   = (float*)alloc((size_t)TT * HH, 4);
    float*  ksum   = (float*)alloc((size_t)TT, 4);

    auto cvt = [&](const float* s, __bf16* d, int n) {
        cvt_f32_bf16<<<(n + 255) / 256, 256, 0, stream>>>(s, d, n);
    };
    cvt(hid, hid_h, TT * EE);
    cvt(Wq,  Wq_h,  EE * EE);
    cvt(Wk,  Wk_h,  LL * EE);
    cvt(Wv,  Wv_h,  LL * EE);
    cvt(Wo,  Wo_h,  EE * EE);
    cvt_qtl<<<(HH * DD * LL + 255) / 256, 256, 0, stream>>>(q2l, qtl_t);
    cvt_vup<<<(HH * LL * DD + 255) / 256, 256, 0, stream>>>(vup, vup_t);

    auto gemm = [&](int M, int N, int K,
                    const __bf16* A, int lda, long long sA,
                    const __bf16* B, int ldb, long long sB,
                    float* Cf, __bf16* Ch, int ldc, long long sC,
                    int transC, int batch) {
        int tiles = ((M + 63) / 64) * ((N + 31) / 32);
        dim3 g((tiles + 7) / 8, batch);
        gemm_nt_bf16<<<g, 256, 0, stream>>>(M, N, K, A, lda, sA, B, ldb, sB,
                                            Cf, Ch, ldc, sC, transC);
    };

    // Q = hidden @ Wq^T  -> [T, E] bf16
    gemm(TT, EE, EE, hid_h, EE, 0, Wq_h, EE, 0, nullptr, Q_h, EE, 0, 0, 1);
    // latent_k / latent_v = hidden @ Wk^T / Wv^T -> [T, L] bf16
    gemm(TT, LL, EE, hid_h, EE, 0, Wk_h, EE, 0, nullptr, lk_h, LL, 0, 0, 1);
    gemm(TT, LL, EE, hid_h, EE, 0, Wv_h, EE, 0, nullptr, lv_h, LL, 0, 0, 1);
    // q_latent[t,h,l] = sum_d Q[t,h,d] * qtl_t[h,l,d]  (batched over heads)
    gemm(TT, LL, DD, Q_h, EE, DD, qtl_t, DD, (long long)LL * DD,
         nullptr, qlat_h, HH * LL, LL, 0, HH);
    // Vt[h,d,t] = sum_l latent_v[t,l] * vup_t[h,d,l]  (store transposed)
    gemm(TT, DD, LL, lv_h, LL, 0, vup_t, LL, (long long)DD * LL,
         nullptr, Vt_h, TT, (long long)DD * TT, 1, HH);

    rowsums<<<(TT * HH + 255) / 256, 256, 0, stream>>>(qlat_h, lk_h, qsum, ksum);

    attn_kernel<<<dim3(TT / 16 / 4, HH), 128, 0, stream>>>(qlat_h, lk_h, Vt_h,
                                                           qsum, ksum, ctx_h);

    // out = ctx @ Wo^T -> f32 d_out
    gemm(TT, EE, EE, ctx_h, EE, 0, Wo_h, EE, 0, out, nullptr, EE, 0, 0, 1);
}
